// SelfCorrectionLoop_23270132810190
// MI455X (gfx1250) — matrix operand order repaired and checked by reading, not compiled, not run on hardware
//
#include <hip/hip_runtime.h>
#include <hip/hip_bf16.h>
#include <stdint.h>

// ---------------------------------------------------------------- constants
#define B_      16
#define T_      512
#define D_      1024
#define H_      4
#define DH_     256
#define L_      2
#define FF_     4096
#define DHALF_  512
#define MAXCORR_ 3
#define THRESH_ 0.3f
#define EPS_    1e-5f

typedef __bf16 bf16_t;
typedef __attribute__((ext_vector_type(16))) __bf16 v16bf;
typedef __attribute__((ext_vector_type(8)))  __bf16 v8bf;
typedef __attribute__((ext_vector_type(8)))  float  v8f;

#define BM 128
#define BN 128
#define BK 32
#define LDK (BK + 8)   // 40 bf16 = 80B row stride: 16B aligned, conflict-free

#define STAGE_A_BYTES (BM * LDK * 2)               // 10240
#define STAGE_B_BYTES (BN * LDK * 2)               // 10240
#define STAGE_BYTES   (STAGE_A_BYTES + STAGE_B_BYTES)

// ---------------------------------------------------------------- WMMA GEMM
// C[M,N] = alpha * A[M,K] * op(B) + bias[n]
//   BT=true : B is [N,K] row-major (Linear weight / K-matrix)  -> A*B^T
//   BT=false: B is [K,N] row-major (V-matrix)                  -> A*B
//   ASY=true: M%BM==0 && N%BN==0 -> unguarded, double-buffered
//             global_load_async_to_lds_b128 staging (ASYNCcnt), peeled tail.
// 8 waves in a 2x4 grid; each wave owns a 64x32 tile = 4x2 WMMA tiles.
// Batched via grid.z: z -> (zo=z/hdiv, zi=z%hdiv).
// C strides are in elements of the output type.
template <bool BT, bool RELU, bool OBF16, bool ASY>
__launch_bounds__(256)
__global__ void gemm_bf16_kernel(const bf16_t* __restrict__ A,
                                 const bf16_t* __restrict__ Bm,
                                 const float*  __restrict__ bias,
                                 void* __restrict__ Cp,
                                 int M, int N, int K,
                                 int lda, int ldb, int ldc, float alpha,
                                 int hdiv,
                                 long sAo, long sAi, long sBo, long sBi,
                                 long sCo, long sCi)
{
    __shared__ __align__(16) char smem[2 * STAGE_BYTES];

    const int z  = blockIdx.z;
    const int zo = z / hdiv;
    const int zi = z % hdiv;
    A  += (size_t)zo * sAo + (size_t)zi * sAi;
    Bm += (size_t)zo * sBo + (size_t)zi * sBi;
    const long coff = (long)zo * sCo + (long)zi * sCi;

    const int tid  = threadIdx.x;
    const int lane = tid & 31;
    const int wave = tid >> 5;          // 0..7
    const int wm   = wave & 1;          // 0..1  (rows of 64)
    const int wn   = wave >> 1;         // 0..3  (cols of 32)
    const int lr   = lane & 15;
    const int hi   = lane >> 4;

    const int m0 = blockIdx.x * BM;
    const int n0 = blockIdx.y * BN;

    v8f acc[4][2] = {};

    // ---- fragment loads per documented CDNA5 VGPR layouts + 8x WMMA
    auto compute_tile = [&](const bf16_t* As, const bf16_t* Bs) {
        v16bf afrag[4], bfrag[2];
#pragma unroll
        for (int i = 0; i < 4; ++i) {
            int r  = wm * 64 + i * 16 + lr;
            int kb = hi * 8;   // lanes<16: K{0..7,16..23}; lanes>=16: K{8..15,24..31}
            union { v16bf v; v8bf h[2]; } u;
            u.h[0] = *(const v8bf*)&As[r * LDK + kb];
            u.h[1] = *(const v8bf*)&As[r * LDK + 16 + kb];
            afrag[i] = u.v;
        }
#pragma unroll
        for (int j = 0; j < 2; ++j) {
            int n  = wn * 32 + j * 16 + lr;
            int kb = hi * 8;   // lanes<16: K{0..7,16..23}; lanes>=16: K{8..15,24..31}
            union { v16bf v; v8bf h[2]; } u;
            u.h[0] = *(const v8bf*)&Bs[n * LDK + kb];
            u.h[1] = *(const v8bf*)&Bs[n * LDK + 16 + kb];
            bfrag[j] = u.v;
        }
#pragma unroll
        for (int i = 0; i < 4; ++i)
#pragma unroll
            for (int j = 0; j < 2; ++j)
                acc[i][j] = __builtin_amdgcn_wmma_f32_16x16x32_bf16(
                    false, afrag[i], false, bfrag[j],
                    (short)0, acc[i][j], false, false);
    };

    if (ASY) {
        // ======== async double-buffered path (CDNA5 ASYNCcnt pipeline) ========
        const unsigned smem_u32 = (unsigned)(size_t)(void*)smem;
        const unsigned long long Abase = (unsigned long long)A;
        const unsigned long long Bbase = (unsigned long long)Bm;

        // 4 async instructions per wave per stage:
        //   A tile 128x32 bf16 -> 2 x (16B/lane), B tile 128x32 -> 2 x (16B/lane)
        auto stage_async = [&](int buf, int kk) {
            const unsigned sbase = smem_u32 + (unsigned)(buf * STAGE_BYTES);
#pragma unroll
            for (int it = 0; it < 2; ++it) {
                int idx = tid + it * 256;
                int r   = idx >> 2;
                int kg  = (idx & 3) << 3;
                unsigned ldsa = sbase + (unsigned)((r * LDK + kg) * 2);
                unsigned goff = (unsigned)(((size_t)(m0 + r) * lda + kk + kg) * 2);
                asm volatile("global_load_async_to_lds_b128 %0, %1, %2"
                             :
                             : "v"(ldsa), "v"(goff), "s"(Abase)
                             : "memory");
            }
#pragma unroll
            for (int it = 0; it < 2; ++it) {
                int idx = tid + it * 256;
                int n   = idx >> 2;
                int kg  = (idx & 3) << 3;
                unsigned ldsb = sbase + (unsigned)(STAGE_A_BYTES + (n * LDK + kg) * 2);
                unsigned goff = (unsigned)(((size_t)(n0 + n) * ldb + kk + kg) * 2);
                asm volatile("global_load_async_to_lds_b128 %0, %1, %2"
                             :
                             : "v"(ldsb), "v"(goff), "s"(Bbase)
                             : "memory");
            }
        };

        stage_async(0, 0);
        int buf = 0;
        int kk  = 0;
        // steady state: branch-free body
        for (; kk < K - BK; kk += BK) {
            stage_async(buf ^ 1, kk + BK);
            // async loads complete in order: <=4 outstanding means the
            // 4 ops of the *current* stage have landed in LDS
            asm volatile("s_wait_asynccnt 0x4" ::: "memory");
            __syncthreads();
            compute_tile((const bf16_t*)(smem + buf * STAGE_BYTES),
                         (const bf16_t*)(smem + buf * STAGE_BYTES + STAGE_A_BYTES));
            __syncthreads();            // all waves done reading before overwrite
            buf ^= 1;
        }
        // peeled tail
        asm volatile("s_wait_asynccnt 0x0" ::: "memory");
        __syncthreads();
        compute_tile((const bf16_t*)(smem + buf * STAGE_BYTES),
                     (const bf16_t*)(smem + buf * STAGE_BYTES + STAGE_A_BYTES));
    } else {
        // ======== generic guarded path (small / NN GEMMs) ========
        bf16_t* As = (bf16_t*)smem;
        bf16_t* Bs = (bf16_t*)(smem + STAGE_A_BYTES);
        for (int kk = 0; kk < K; kk += BK) {
#pragma unroll
            for (int it = 0; it < 2; ++it) {
                int idx = tid + it * 256;
                int r   = idx >> 2;
                int kg  = (idx & 3) << 3;
                int gr  = m0 + r;
                uint4 v{0u, 0u, 0u, 0u};
                if (gr < M)
                    v = *(const uint4*)(A + (size_t)gr * lda + kk + kg);
                *(uint4*)&As[r * LDK + kg] = v;
            }
            if (BT) {
#pragma unroll
                for (int it = 0; it < 2; ++it) {
                    int idx = tid + it * 256;
                    int n   = idx >> 2;
                    int kg  = (idx & 3) << 3;
                    int gn  = n0 + n;
                    uint4 v{0u, 0u, 0u, 0u};
                    if (gn < N)
                        v = *(const uint4*)(Bm + (size_t)gn * ldb + kk + kg);
                    *(uint4*)&Bs[n * LDK + kg] = v;
                }
            } else {
#pragma unroll
                for (int it = 0; it < 2; ++it) {
                    int k  = tid >> 3;                          // 0..31
                    int ng = (((tid & 7) | (it << 3)) << 3);    // 0..120
                    uint4 v{0u, 0u, 0u, 0u};
                    int gn = n0 + ng;
                    if (gn < N)
                        v = *(const uint4*)(Bm + (size_t)(kk + k) * ldb + gn);
                    const unsigned short* pv = (const unsigned short*)&v;
#pragma unroll
                    for (int j = 0; j < 8; ++j)
                        *(unsigned short*)&Bs[(ng + j) * LDK + k] = pv[j];
                }
            }
            __syncthreads();
            compute_tile(As, Bs);
            __syncthreads();
        }
    }

    // ---- epilogue: lane<16 -> N=lane, M=vgpr; lane>=16 -> N=lane-16, M=8+vgpr
    float*  Cf = (float*)Cp + coff;
    bf16_t* Ch = (bf16_t*)Cp + coff;
#pragma unroll
    for (int i = 0; i < 4; ++i) {
#pragma unroll
        for (int j = 0; j < 2; ++j) {
            int col   = n0 + wn * 32 + j * 16 + lr;
            int rbase = m0 + wm * 64 + i * 16 + hi * 8;
            float bv  = (bias != nullptr && col < N) ? bias[col] : 0.f;
#pragma unroll
            for (int v = 0; v < 8; ++v) {
                int r = rbase + v;
                if (r < M && col < N) {
                    float val = acc[i][j][v] * alpha + bv;
                    if (RELU) val = val > 0.f ? val : 0.f;
                    if (OBF16) Ch[(size_t)r * ldc + col] = (bf16_t)val;
                    else       Cf[(size_t)r * ldc + col] = val;
                }
            }
        }
    }
}

// ---------------------------------------------------------------- misc kernels
__global__ void k_f2b(const float* __restrict__ x, bf16_t* __restrict__ y, long n)
{
    long i = (long)blockIdx.x * blockDim.x + threadIdx.x;
    long s = (long)gridDim.x * blockDim.x;
    for (; i < n; i += s) y[i] = (bf16_t)x[i];
}

__global__ void k_copy(const float* __restrict__ x, float* __restrict__ y, long n)
{
    long i = (long)blockIdx.x * blockDim.x + threadIdx.x;
    long s = (long)gridDim.x * blockDim.x;
    for (; i < n; i += s) y[i] = x[i];
}

__global__ void k_zero_i(int* p, int n)
{
    int i = blockIdx.x * blockDim.x + threadIdx.x;
    if (i < n) p[i] = 0;
}

// causal softmax over one row of S (fp32) -> P (bf16); grid (T, B*H), 256 threads
__launch_bounds__(256)
__global__ void k_softmax(const float* __restrict__ S, bf16_t* __restrict__ P)
{
    const int row = blockIdx.x;
    const int bh  = blockIdx.y;
    const float* s = S + ((size_t)bh * T_ + row) * T_;
    bf16_t*      p = P + ((size_t)bh * T_ + row) * T_;
    const int tid = threadIdx.x;
    __shared__ float red[256];

    float v0 = (tid       <= row) ? s[tid]       : -1e30f;
    float v1 = (tid + 256 <= row) ? s[tid + 256] : -1e30f;
    red[tid] = fmaxf(v0, v1);
    __syncthreads();
    for (int o = 128; o > 0; o >>= 1) {
        if (tid < o) red[tid] = fmaxf(red[tid], red[tid + o]);
        __syncthreads();
    }
    float m = red[0];
    __syncthreads();
    float e0 = (tid       <= row) ? __expf(v0 - m) : 0.f;
    float e1 = (tid + 256 <= row) ? __expf(v1 - m) : 0.f;
    red[tid] = e0 + e1;
    __syncthreads();
    for (int o = 128; o > 0; o >>= 1) {
        if (tid < o) red[tid] += red[tid + o];
        __syncthreads();
    }
    float inv = 1.f / red[0];
    p[tid]       = (bf16_t)(e0 * inv);
    p[tid + 256] = (bf16_t)(e1 * inv);
}

// out = LayerNorm(X + R) * g + b ; grid B*T, 256 threads, D=1024 (4/thread)
__launch_bounds__(256)
__global__ void k_ln_res(const float* __restrict__ X, const float* __restrict__ R,
                         const float* __restrict__ g, const float* __restrict__ bb,
                         float* __restrict__ out)
{
    const long row = blockIdx.x;
    const float* x = X + row * D_;
    const float* r = R + row * D_;
    float*       o = out + row * D_;
    const int tid = threadIdx.x;
    __shared__ float red[256];

    float v[4], sum = 0.f;
#pragma unroll
    for (int i = 0; i < 4; ++i) { v[i] = x[tid + i * 256] + r[tid + i * 256]; sum += v[i]; }
    red[tid] = sum;
    __syncthreads();
    for (int s = 128; s > 0; s >>= 1) { if (tid < s) red[tid] += red[tid + s]; __syncthreads(); }
    float mu = red[0] * (1.f / D_);
    __syncthreads();
    float sq = 0.f;
#pragma unroll
    for (int i = 0; i < 4; ++i) { float d = v[i] - mu; sq += d * d; }
    red[tid] = sq;
    __syncthreads();
    for (int s = 128; s > 0; s >>= 1) { if (tid < s) red[tid] += red[tid + s]; __syncthreads(); }
    float rstd = rsqrtf(red[0] * (1.f / D_) + EPS_);
#pragma unroll
    for (int i = 0; i < 4; ++i) {
        int c = tid + i * 256;
        o[c] = (v[i] - mu) * rstd * g[c] + bb[c];
    }
}

// scores[row] = sigmoid(dot(Hrow(bf16,512), w2(f32)) + b2); grid B*T
__launch_bounds__(256)
__global__ void k_head(const bf16_t* __restrict__ Hb, const float* __restrict__ w2,
                       const float* __restrict__ b2, float* __restrict__ sc)
{
    const long row = blockIdx.x;
    const bf16_t* h = Hb + row * DHALF_;
    const int tid = threadIdx.x;
    __shared__ float red[256];
    red[tid] = (float)h[tid] * w2[tid] + (float)h[tid + 256] * w2[tid + 256];
    __syncthreads();
    for (int s = 128; s > 0; s >>= 1) { if (tid < s) red[tid] += red[tid + s]; __syncthreads(); }
    if (tid == 0) sc[row] = 1.f / (1.f + __expf(-(red[0] + b2[0])));
}

// first argmin among below-threshold steps; grid B
__launch_bounds__(256)
__global__ void k_worst(const float* __restrict__ sc, int* __restrict__ w, int* __restrict__ hb)
{
    const int b = blockIdx.x;
    const int tid = threadIdx.x;
    const float* s = sc + b * T_;
    float bestv = 1e38f;
    int   besti = T_;
    for (int j = tid; j < T_; j += 256) {
        float v   = s[j];
        float key = (v < THRESH_) ? v : 1e38f;
        if (key < bestv || (key == bestv && j < besti)) { bestv = key; besti = j; }
    }
    __shared__ float rv[256];
    __shared__ int   ri[256];
    rv[tid] = bestv; ri[tid] = besti;
    __syncthreads();
    for (int o = 128; o > 0; o >>= 1) {
        if (tid < o) {
            float v2 = rv[tid + o]; int i2 = ri[tid + o];
            if (v2 < rv[tid] || (v2 == rv[tid] && i2 < ri[tid])) { rv[tid] = v2; ri[tid] = i2; }
        }
        __syncthreads();
    }
    if (tid == 0) {
        int idx = ri[0];
        if (idx >= T_) idx = 0;          // all-inf -> argmin = 0 (matches jnp)
        w[b]  = idx;
        hb[b] = (rv[0] < 1e37f) ? 1 : 0;
    }
}

// build [cur, ctx] (bf16) for correction MLP; grid B
__launch_bounds__(256)
__global__ void k_ctx(const float* __restrict__ CORR, const int* __restrict__ w,
                      bf16_t* __restrict__ cinpb)
{
    const int b = blockIdx.x;
    const int tid = threadIdx.x;
    const int wi = w[b];
    const int i0 = wi - 1, i2 = wi + 1;
    const float cnt = 1.f + (i0 >= 0 ? 1.f : 0.f) + (i2 < T_ ? 1.f : 0.f);
    const float* base = CORR + (size_t)b * T_ * D_;
    bf16_t* out = cinpb + (size_t)b * 2 * D_;
    for (int d = tid; d < D_; d += 256) {
        float cur = base[(size_t)wi * D_ + d];
        float s = cur;
        if (i0 >= 0) s += base[(size_t)i0 * D_ + d];
        if (i2 < T_) s += base[(size_t)i2 * D_ + d];
        out[d]      = (bf16_t)cur;
        out[D_ + d] = (bf16_t)(s / cnt);
    }
}

// apply correction + increment counts; grid B
__launch_bounds__(256)
__global__ void k_apply(float* __restrict__ CORR, const float* __restrict__ ns,
                        const int* __restrict__ w, const int* __restrict__ hb,
                        int* __restrict__ counts)
{
    const int b = blockIdx.x;
    const int tid = threadIdx.x;
    if (hb[b]) {
        const int wi = w[b];
        for (int d = tid; d < D_; d += 256)
            CORR[((size_t)b * T_ + wi) * D_ + d] = ns[b * D_ + d];
        if (tid == 0) counts[b] += 1;
    }
}

// means + improved + counts as float; grid B
__launch_bounds__(256)
__global__ void k_final(const float* __restrict__ sc0, const float* __restrict__ scf,
                        const int* __restrict__ counts,
                        float* __restrict__ outc, float* __restrict__ outim)
{
    const int b = blockIdx.x;
    const int tid = threadIdx.x;
    __shared__ float r0[256], r1[256];
    r0[tid] = sc0[b * T_ + tid] + sc0[b * T_ + tid + 256];
    r1[tid] = scf[b * T_ + tid] + scf[b * T_ + tid + 256];
    __syncthreads();
    for (int o = 128; o > 0; o >>= 1) {
        if (tid < o) { r0[tid] += r0[tid + o]; r1[tid] += r1[tid + o]; }
        __syncthreads();
    }
    if (tid == 0) {
        outc[b]  = (float)counts[b];
        outim[b] = (r1[0] > r0[0]) ? 1.f : 0.f;
    }
}

// ---------------------------------------------------------------- host helpers
struct GemmDesc {
    const bf16_t* A; const bf16_t* B; const float* bias; void* C;
    int M, N, K, lda, ldb, ldc; float alpha;
    int bz, hdiv;
    long sAo, sAi, sBo, sBi, sCo, sCi;
};

enum GemmMode { NT_F32 = 0, NT_BF16 = 1, NT_RELU_BF16 = 2, NN_F32 = 3 };

static void gemm_launch(hipStream_t st, const GemmDesc& g, int mode)
{
    dim3 grid((g.M + BM - 1) / BM, (g.N + BN - 1) / BN, g.bz);
    dim3 blk(256);
    const bool asy = (mode != NN_F32) && (g.M % BM == 0) && (g.N % BN == 0);
#define GEMM_ARGS g.A, g.B, g.bias, g.C, g.M, g.N, g.K, g.lda, g.ldb, g.ldc, \
                  g.alpha, g.hdiv, g.sAo, g.sAi, g.sBo, g.sBi, g.sCo, g.sCi
    switch (mode) {
    case NT_F32:
        if (asy) gemm_bf16_kernel<true, false, false, true ><<<grid, blk, 0, st>>>(GEMM_ARGS);
        else     gemm_bf16_kernel<true, false, false, false><<<grid, blk, 0, st>>>(GEMM_ARGS);
        break;
    case NT_BF16:
        if (asy) gemm_bf16_kernel<true, false, true, true ><<<grid, blk, 0, st>>>(GEMM_ARGS);
        else     gemm_bf16_kernel<true, false, true, false><<<grid, blk, 0, st>>>(GEMM_ARGS);
        break;
    case NT_RELU_BF16:
        if (asy) gemm_bf16_kernel<true, true, true, true ><<<grid, blk, 0, st>>>(GEMM_ARGS);
        else     gemm_bf16_kernel<true, true, true, false><<<grid, blk, 0, st>>>(GEMM_ARGS);
        break;
    default:
        gemm_bf16_kernel<false, false, false, false><<<grid, blk, 0, st>>>(GEMM_ARGS);
        break;
    }
#undef GEMM_ARGS
}

static GemmDesc mk(const bf16_t* A, const bf16_t* B, const float* bias, void* C,
                   int M, int N, int K, int lda, int ldb, int ldc, float alpha)
{
    GemmDesc g{};
    g.A = A; g.B = B; g.bias = bias; g.C = C;
    g.M = M; g.N = N; g.K = K; g.lda = lda; g.ldb = ldb; g.ldc = ldc;
    g.alpha = alpha; g.bz = 1; g.hdiv = 1;
    g.sAo = g.sAi = g.sBo = g.sBi = g.sCo = g.sCi = 0;
    return g;
}

// Pointers carved from the workspace
struct Ws {
    bf16_t *wq, *wo, *w1, *w2, *rh1, *cw1, *cw2;
    float  *X, *CORR, *OB, *Y2, *NS;
    bf16_t *ABUF, *HBUF, *QKV, *P, *CINB;
    float  *S, *SC0, *SC, *SCF;
    int    *widx, *hb, *counts;
};

struct InPtrs {
    const float *chain, *Wqkv, *bqkv, *Wo, *bo, *ln1_g, *ln1_b, *W1, *b1, *W2, *b2,
                *ln2_g, *ln2_b, *rh_W1, *rh_b1, *rh_W2, *rh_b2,
                *cor_W1, *cor_b1, *cor_W2, *cor_b2;
};

static inline void cvt(hipStream_t st, const float* src, bf16_t* dst, long n)
{
    int blocks = (int)((n + 1023) / 1024);
    if (blocks > 4096) blocks = 4096;
    k_f2b<<<blocks, 256, 0, st>>>(src, dst, n);
}

static inline void cpyf(hipStream_t st, const float* src, float* dst, long n)
{
    int blocks = (int)((n + 1023) / 1024);
    if (blocks > 4096) blocks = 4096;
    k_copy<<<blocks, 256, 0, st>>>(src, dst, n);
}

// one transformer+head pass: src (fp32 B*T*D) -> scores (fp32 B*T)
static void run_prm(hipStream_t st, const float* src, float* scores,
                    const Ws& w, const InPtrs& in)
{
    const long BT  = (long)B_ * T_;
    const long BTD = BT * D_;
    const int  D3  = 3 * D_;

    cpyf(st, src, w.X, BTD);

    for (int l = 0; l < L_; ++l) {
        // QKV projection -> bf16 packed [B*T, 3D]
        cvt(st, w.X, w.ABUF, BTD);
        gemm_launch(st,
            mk(w.ABUF, w.wq + (size_t)l * D3 * D_, in.bqkv + (size_t)l * D3,
               w.QKV, (int)BT, D3, D_, D_, D_, D3, 1.f),
            NT_BF16);

        // scores = Q K^T / sqrt(DH), batched over (b,h)
        {
            GemmDesc g = mk(w.QKV, w.QKV + D_, nullptr, w.S,
                            T_, T_, DH_, D3, D3, T_, 0.0625f);
            g.bz = B_ * H_; g.hdiv = H_;
            g.sAo = (long)T_ * D3;  g.sAi = DH_;
            g.sBo = (long)T_ * D3;  g.sBi = DH_;
            g.sCo = (long)H_ * T_ * T_; g.sCi = (long)T_ * T_;
            gemm_launch(st, g, NT_F32);
        }

        // causal softmax -> bf16 probs
        {
            dim3 grid(T_, B_ * H_);
            k_softmax<<<grid, 256, 0, st>>>(w.S, w.P);
        }

        // O = P V, batched over (b,h), NN
        {
            GemmDesc g = mk(w.P, w.QKV + 2 * D_, nullptr, w.OB,
                            T_, DH_, T_, T_, D3, D_, 1.f);
            g.bz = B_ * H_; g.hdiv = H_;
            g.sAo = (long)H_ * T_ * T_; g.sAi = (long)T_ * T_;
            g.sBo = (long)T_ * D3;      g.sBi = DH_;
            g.sCo = (long)T_ * D_;      g.sCi = DH_;
            gemm_launch(st, g, NN_F32);
        }

        // output projection
        cvt(st, w.OB, w.ABUF, BTD);
        gemm_launch(st,
            mk(w.ABUF, w.wo + (size_t)l * D_ * D_, in.bo + (size_t)l * D_,
               w.Y2, (int)BT, D_, D_, D_, D_, D_, 1.f),
            NT_F32);
        k_ln_res<<<(int)BT, 256, 0, st>>>(w.X, w.Y2,
            in.ln1_g + (size_t)l * D_, in.ln1_b + (size_t)l * D_, w.X);

        // feed-forward
        cvt(st, w.X, w.ABUF, BTD);
        gemm_launch(st,
            mk(w.ABUF, w.w1 + (size_t)l * FF_ * D_, in.b1 + (size_t)l * FF_,
               w.HBUF, (int)BT, FF_, D_, D_, D_, FF_, 1.f),
            NT_RELU_BF16);
        gemm_launch(st,
            mk(w.HBUF, w.w2 + (size_t)l * D_ * FF_, in.b2 + (size_t)l * D_,
               w.Y2, (int)BT, D_, FF_, FF_, FF_, D_, 1.f),
            NT_F32);
        k_ln_res<<<(int)BT, 256, 0, st>>>(w.X, w.Y2,
            in.ln2_g + (size_t)l * D_, in.ln2_b + (size_t)l * D_, w.X);
    }

    // regression head
    cvt(st, w.X, w.ABUF, BTD);
    gemm_launch(st,
        mk(w.ABUF, w.rh1, in.rh_b1, w.HBUF,
           (int)BT, DHALF_, D_, D_, D_, DHALF_, 1.f),
        NT_RELU_BF16);
    k_head<<<(int)BT, 256, 0, st>>>(w.HBUF, in.rh_W2, in.rh_b2, scores);
}

// ---------------------------------------------------------------- entry point
extern "C" void kernel_launch(void* const* d_in, const int* in_sizes, int n_in,
                              void* d_out, int out_size, void* d_ws, size_t ws_size,
                              hipStream_t stream)
{
    (void)in_sizes; (void)n_in; (void)out_size; (void)ws_size;

    InPtrs in;
    in.chain  = (const float*)d_in[0];
    in.Wqkv   = (const float*)d_in[1];
    in.bqkv   = (const float*)d_in[2];
    in.Wo     = (const float*)d_in[3];
    in.bo     = (const float*)d_in[4];
    in.ln1_g  = (const float*)d_in[5];
    in.ln1_b  = (const float*)d_in[6];
    in.W1     = (const float*)d_in[7];
    in.b1     = (const float*)d_in[8];
    in.W2     = (const float*)d_in[9];
    in.b2     = (const float*)d_in[10];
    in.ln2_g  = (const float*)d_in[11];
    in.ln2_b  = (const float*)d_in[12];
    in.rh_W1  = (const float*)d_in[13];
    in.rh_b1  = (const float*)d_in[14];
    in.rh_W2  = (const float*)d_in[15];
    in.rh_b2  = (const float*)d_in[16];
    in.cor_W1 = (const float*)d_in[17];
    in.cor_b1 = (const float*)d_in[18];
    in.cor_W2 = (const float*)d_in[19];
    in.cor_b2 = (const float*)d_in[20];

    const long BT  = (long)B_ * T_;
    const long BTD = BT * D_;
    const int  D3  = 3 * D_;

    // ---- carve workspace
    char* base = (char*)d_ws;
    size_t off = 0;
    auto carve = [&](size_t bytes) -> void* {
        void* p = base + off;
        off += (bytes + 255) & ~(size_t)255;
        return p;
    };

    Ws w;
    w.wq   = (bf16_t*)carve((size_t)L_ * D3 * D_  * sizeof(bf16_t));
    w.wo   = (bf16_t*)carve((size_t)L_ * D_ * D_  * sizeof(bf16_t));
    w.w1   = (bf16_t*)carve((size_t)L_ * FF_ * D_ * sizeof(bf16_t));
    w.w2   = (bf16_t*)carve((size_t)L_ * D_ * FF_ * sizeof(bf16_t));
    w.rh1  = (bf16_t*)carve((size_t)DHALF_ * D_   * sizeof(bf16_t));
    w.cw1  = (bf16_t*)carve((size_t)2 * D_ * 2 * D_ * sizeof(bf16_t));
    w.cw2  = (bf16_t*)carve((size_t)D_ * 2 * D_   * sizeof(bf16_t));

    w.X    = (float*)carve(BTD * sizeof(float));
    w.CORR = (float*)carve(BTD * sizeof(float));
    w.OB   = (float*)carve(BTD * sizeof(float));
    w.Y2   = (float*)carve(BTD * sizeof(float));
    w.NS   = (float*)carve((size_t)B_ * D_ * sizeof(float));

    w.ABUF = (bf16_t*)carve((size_t)BT * FF_ * sizeof(bf16_t));
    w.HBUF = (bf16_t*)carve((size_t)BT * FF_ * sizeof(bf16_t));
    w.QKV  = (bf16_t*)carve((size_t)BT * D3  * sizeof(bf16_t));
    w.P    = (bf16_t*)carve((size_t)B_ * H_ * T_ * T_ * sizeof(bf16_t));
    w.CINB = (bf16_t*)carve((size_t)B_ * 2 * D_ * sizeof(bf16_t));

    w.S    = (float*)carve((size_t)B_ * H_ * T_ * T_ * sizeof(float));
    w.SC0  = (float*)carve(BT * sizeof(float));
    w.SC   = (float*)carve(BT * sizeof(float));
    w.SCF  = (float*)carve(BT * sizeof(float));

    w.widx   = (int*)carve(B_ * sizeof(int));
    w.hb     = (int*)carve(B_ * sizeof(int));
    w.counts = (int*)carve(B_ * sizeof(int));

    // ---- one-time-per-launch setup (deterministic, replay-safe)
    k_zero_i<<<1, 32, 0, stream>>>(w.counts, B_);
    cpyf(stream, in.chain, w.CORR, BTD);

    cvt(stream, in.Wqkv,   w.wq,  (long)L_ * D3 * D_);
    cvt(stream, in.Wo,     w.wo,  (long)L_ * D_ * D_);
    cvt(stream, in.W1,     w.w1,  (long)L_ * FF_ * D_);
    cvt(stream, in.W2,     w.w2,  (long)L_ * D_ * FF_);
    cvt(stream, in.rh_W1,  w.rh1, (long)DHALF_ * D_);
    cvt(stream, in.cor_W1, w.cw1, (long)2 * D_ * 2 * D_);
    cvt(stream, in.cor_W2, w.cw2, (long)D_ * 2 * D_);

    // ---- scores0 = prm(chain)  (== first loop iteration's s, since corrected==chain)
    run_prm(stream, in.chain, w.SC0, w, in);

    // ---- MAX_CORR correction iterations
    for (int it = 0; it < MAXCORR_; ++it) {
        const float* s = w.SC;
        if (it == 0) {
            s = w.SC0;                       // prm(corrected)==prm(chain) on iter 0
        } else {
            run_prm(stream, w.CORR, w.SC, w, in);
        }
        k_worst<<<B_, 256, 0, stream>>>(s, w.widx, w.hb);
        k_ctx<<<B_, 256, 0, stream>>>(w.CORR, w.widx, w.CINB);
        gemm_launch(stream,
            mk(w.CINB, w.cw1, in.cor_b1, w.HBUF,
               B_, 2 * D_, 2 * D_, 2 * D_, 2 * D_, 2 * D_, 1.f),
            NT_RELU_BF16);
        gemm_launch(stream,
            mk(w.HBUF, w.cw2, in.cor_b2, w.NS,
               B_, D_, 2 * D_, 2 * D_, 2 * D_, D_, 1.f),
            NT_F32);
        k_apply<<<B_, 256, 0, stream>>>(w.CORR, w.NS, w.widx, w.hb, w.counts);
    }

    // ---- final scores + outputs
    run_prm(stream, w.CORR, w.SCF, w, in);

    float* out = (float*)d_out;
    float* out_corr     = out;                      // (B,T,D)
    float* out_counts   = out + BTD;                // (B,)
    float* out_final    = out + BTD + B_;           // (B,T)
    float* out_improved = out + BTD + B_ + BT;      // (B,)

    cpyf(stream, w.CORR, out_corr, BTD);
    cpyf(stream, w.SCF, out_final, BT);
    k_final<<<B_, 256, 0, stream>>>(w.SC0, w.SCF, w.counts, out_counts, out_improved);
}
